// LFQQuantizer_25409026523969
// MI455X (gfx1250) — matrix operand order repaired and checked by reading, not compiled, not run on hardware
//
#include <hip/hip_runtime.h>

typedef float  v2f  __attribute__((ext_vector_type(2)));
typedef float  v4f  __attribute__((ext_vector_type(4)));
typedef float  v8f  __attribute__((ext_vector_type(8)));
typedef __bf16 v4bf  __attribute__((ext_vector_type(4)));
typedef __bf16 v8bf  __attribute__((ext_vector_type(8)));
typedef __bf16 v16bf __attribute__((ext_vector_type(16)));

#define NUM_CODES 1024
#define CODE_DIM  256
#define ZSTRIDE   264          // bf16 row pitch: 528B = 33*16B -> aligned, bank-spread
#define BSH_TILE_ELEMS (8 * 32 * 16)   // per 16-code tile: 8 K-steps * 32 lanes * 16 bf16

// ---------------------------------------------------------------------------
// Prep 1: c_sq[k] = sum_d codebook[k][d]^2   (one wave per code, fp32)
// ---------------------------------------------------------------------------
__global__ __launch_bounds__(256) void vq_csq_kernel(const float* __restrict__ cb,
                                                     float* __restrict__ csq) {
  int code = blockIdx.x * 8 + (threadIdx.x >> 5);
  int lane = threadIdx.x & 31;
  const float* row = cb + (size_t)code * CODE_DIM;
  float s = 0.0f;
  #pragma unroll
  for (int d = lane; d < CODE_DIM; d += 32) {
    float x = row[d];
    s += x * x;
  }
  #pragma unroll
  for (int off = 16; off >= 1; off >>= 1) s += __shfl_xor(s, off, 32);
  if (lane == 0) csq[code] = s;
}

// ---------------------------------------------------------------------------
// Prep 2: split codebook into bf16 hi/lo, pre-shuffled into this kernel's B-fragment
// layout for 16x16x32: lane = {n = lane&15, Khalf = lane>>4}; each lane owns 16
// contiguous K values => one contiguous 32B chunk per (tile, kstep, lane).
// ---------------------------------------------------------------------------
__global__ __launch_bounds__(256) void vq_shufbf_kernel(const float* __restrict__ cb,
                                                        unsigned short* __restrict__ bH_,
                                                        unsigned short* __restrict__ bL_) {
  __bf16* bH = (__bf16*)bH_;
  __bf16* bL = (__bf16*)bL_;
  int f    = blockIdx.x * 256 + threadIdx.x;   // 0 .. 64*8*32-1
  int lane = f & 31;
  int kk   = (f >> 5) & 7;
  int tile = f >> 8;
  int n    = lane & 15;
  int h    = lane >> 4;
  int K0   = kk * 32 + h * 16;
  const float* src = cb + (size_t)(tile * 16 + n) * CODE_DIM + K0;
  size_t dst = (size_t)f * 16;
  #pragma unroll
  for (int i = 0; i < 16; ++i) {
    float x = src[i];
    __bf16 hi = (__bf16)x;
    __bf16 lo = (__bf16)(x - (float)hi);
    bH[dst + i] = hi;
    bL[dst + i] = lo;
  }
}

// ---------------------------------------------------------------------------
// Main (primary): bf16x3 split GEMM via v_wmma_f32_16x16x32_bf16.
// Block = 32 query rows vs all 1024 codes; wave owns 64 codes per pass x 2
// passes. Each (bh,bl) fragment pair is reused by two M-half accumulator
// chains (6 WMMAs per 64B of B data) -> halves L2 traffic + hides XDL latency.
// ---------------------------------------------------------------------------
__global__ __launch_bounds__(256) void vq_main_bf16_kernel(
    const float* __restrict__ zE, const float* __restrict__ cb,
    const unsigned short* __restrict__ bH_, const unsigned short* __restrict__ bL_,
    const float* __restrict__ csq,
    float* __restrict__ outZq, float* __restrict__ outIdx) {
  __shared__ __bf16 sZh[32 * ZSTRIDE];
  __shared__ __bf16 sZl[32 * ZSTRIDE];
  __shared__ float sMinV[8][32];
  __shared__ int   sMinI[8][32];
  __shared__ int   sFinal[32];

  const int tid  = threadIdx.x;
  const int lane = tid & 31;
  const int wave = tid >> 5;
  const int m0   = blockIdx.x * 32;

  // ---- stage 32x256 z tile: f32 -> bf16 hi + bf16 residual, padded pitch ----
  {
    const v4f* gz = (const v4f*)(zE + (size_t)m0 * CODE_DIM);
    #pragma unroll
    for (int i = 0; i < 8; ++i) {
      int q  = tid + i * 256;          // 0..2047 float4 slots
      int m  = q >> 6;
      int dq = q & 63;
      v4f v = gz[q];
      v4bf h, l;
      #pragma unroll
      for (int c = 0; c < 4; ++c) {
        __bf16 hi = (__bf16)v[c];
        h[c] = hi;
        l[c] = (__bf16)(v[c] - (float)hi);
      }
      *(v4bf*)&sZh[m * ZSTRIDE + dq * 4] = h;
      *(v4bf*)&sZl[m * ZSTRIDE + dq * 4] = l;
    }
  }
  __syncthreads();

  const int mLane = lane & 15;
  const int aOff  = (lane >> 4) << 4;    // K-half offset within A chunk

  const __bf16* aH0 = &sZh[mLane * ZSTRIDE + aOff];
  const __bf16* aL0 = &sZl[mLane * ZSTRIDE + aOff];
  const __bf16* aH1 = aH0 + 16 * ZSTRIDE;
  const __bf16* aL1 = aL0 + 16 * ZSTRIDE;
  const __bf16* bH  = (const __bf16*)bH_;
  const __bf16* bL  = (const __bf16*)bL_;
  const size_t bLaneOff = (size_t)lane * 16;

  float bestV[16];                       // [mhalf*8 + r]
  int   bestI[16];
  #pragma unroll
  for (int r = 0; r < 16; ++r) { bestV[r] = 3.4e38f; bestI[r] = 0; }

  #pragma unroll 1
  for (int pass = 0; pass < 2; ++pass) {
    const int n0base = wave * 64 + pass * 512;   // codes ascend across passes

    v8f acc[8];                          // [mhalf*4 + j]
    v8f zero = {};
    #pragma unroll
    for (int j = 0; j < 8; ++j) acc[j] = zero;

    const size_t bTile0 = (size_t)(n0base >> 4) * BSH_TILE_ELEMS;

    for (int kk = 0; kk < 8; ++kk) {     // 8 K-steps of 32 over D=256
      int K = kk * 32;
      v8bf h00 = *(const v8bf*)(aH0 + K);
      v8bf h01 = *(const v8bf*)(aH0 + K + 8);
      v8bf l00 = *(const v8bf*)(aL0 + K);
      v8bf l01 = *(const v8bf*)(aL0 + K + 8);
      v8bf h10 = *(const v8bf*)(aH1 + K);
      v8bf h11 = *(const v8bf*)(aH1 + K + 8);
      v8bf l10 = *(const v8bf*)(aL1 + K);
      v8bf l11 = *(const v8bf*)(aL1 + K + 8);
      v16bf ah0 = __builtin_shufflevector(h00, h01, 0,1,2,3,4,5,6,7,8,9,10,11,12,13,14,15);
      v16bf al0 = __builtin_shufflevector(l00, l01, 0,1,2,3,4,5,6,7,8,9,10,11,12,13,14,15);
      v16bf ah1 = __builtin_shufflevector(h10, h11, 0,1,2,3,4,5,6,7,8,9,10,11,12,13,14,15);
      v16bf al1 = __builtin_shufflevector(l10, l11, 0,1,2,3,4,5,6,7,8,9,10,11,12,13,14,15);

      size_t bk = bTile0 + (size_t)kk * (32 * 16) + bLaneOff;
      #pragma unroll
      for (int j = 0; j < 4; ++j) {
        size_t off = bk + (size_t)j * BSH_TILE_ELEMS;
        v16bf bh = *(const v16bf*)(bH + off);   // 32B contiguous per lane
        v16bf bl = *(const v16bf*)(bL + off);
        // interleave the two M-half chains: consecutive WMMAs are independent
        acc[j]     = __builtin_amdgcn_wmma_f32_16x16x32_bf16(
            false, ah0, false, bh, (short)0, acc[j],     false, false);
        acc[4 + j] = __builtin_amdgcn_wmma_f32_16x16x32_bf16(
            false, ah1, false, bh, (short)0, acc[4 + j], false, false);
        acc[j]     = __builtin_amdgcn_wmma_f32_16x16x32_bf16(
            false, ah0, false, bl, (short)0, acc[j],     false, false);
        acc[4 + j] = __builtin_amdgcn_wmma_f32_16x16x32_bf16(
            false, ah1, false, bl, (short)0, acc[4 + j], false, false);
        acc[j]     = __builtin_amdgcn_wmma_f32_16x16x32_bf16(
            false, al0, false, bh, (short)0, acc[j],     false, false);
        acc[4 + j] = __builtin_amdgcn_wmma_f32_16x16x32_bf16(
            false, al1, false, bh, (short)0, acc[4 + j], false, false);
      }
    }

    // ---- fold this pass into the persistent per-lane argmin state ----
    #pragma unroll
    for (int j = 0; j < 4; ++j) {
      int code = n0base + j * 16 + mLane;
      float cs;
      if (csq) {
        cs = csq[code];
      } else {
        cs = 0.0f;
        const float* crow = cb + (size_t)code * CODE_DIM;
        for (int d = 0; d < CODE_DIM; ++d) cs += crow[d] * crow[d];
      }
      #pragma unroll
      for (int mh = 0; mh < 2; ++mh) {
        #pragma unroll
        for (int r = 0; r < 8; ++r) {
          float s = cs - 2.0f * acc[mh * 4 + j][r];
          int b = mh * 8 + r;
          if (s < bestV[b]) { bestV[b] = s; bestI[b] = code; }
        }
      }
    }
  }

  // ---- reduce across the 16 N-lanes of each half (tie-break: lower index) ----
  #pragma unroll
  for (int b = 0; b < 16; ++b) {
    float v = bestV[b];
    int   ii = bestI[b];
    #pragma unroll
    for (int off = 8; off >= 1; off >>= 1) {
      float ov = __shfl_xor(v, off, 32);
      int   oi = __shfl_xor(ii, off, 32);
      if (ov < v || (ov == v && oi < ii)) { v = ov; ii = oi; }
    }
    if (mLane == 0) {
      int row = (b >> 3) * 16 + (b & 7) + ((lane >> 4) << 3);
      sMinV[wave][row] = v;
      sMinI[wave][row] = ii;
    }
  }
  __syncthreads();

  // ---- cross-wave reduce: 32 rows, one thread each ----
  if (tid < 32) {
    float v = sMinV[0][tid];
    int   ii = sMinI[0][tid];
    #pragma unroll
    for (int w = 1; w < 8; ++w) {
      float ov = sMinV[w][tid];
      int   oi = sMinI[w][tid];
      if (ov < v || (ov == v && oi < ii)) { v = ov; ii = oi; }
    }
    sFinal[tid] = ii;
    outIdx[m0 + tid] = (float)ii;
  }
  __syncthreads();

  // ---- gather z_q = codebook[idx]; wave copies 4 rows, float4 per lane ----
  #pragma unroll
  for (int rr = 0; rr < 4; ++rr) {
    int row  = wave * 4 + rr;
    int code = sFinal[row];
    const v4f* src = (const v4f*)(cb + (size_t)code * CODE_DIM);
    v4f* dst = (v4f*)(outZq + (size_t)(m0 + row) * CODE_DIM);
    dst[lane]      = src[lane];
    dst[lane + 32] = src[lane + 32];
  }
}

// ---------------------------------------------------------------------------
// Fallback (exact fp32): v_wmma_f32_16x16x4_f32, B direct from L2-resident
// codebook. Used only when workspace is too small for the split layout.
// ---------------------------------------------------------------------------
#define F32_LDS_STRIDE 260
__global__ __launch_bounds__(256) void vq_main_f32_kernel(
    const float* __restrict__ zE, const float* __restrict__ cb,
    const float* __restrict__ csq,
    float* __restrict__ outZq, float* __restrict__ outIdx) {
  __shared__ float sZ[16 * F32_LDS_STRIDE];
  __shared__ float sMinV[8][16];
  __shared__ int   sMinI[8][16];
  __shared__ int   sFinal[16];

  const int tid  = threadIdx.x;
  const int lane = tid & 31;
  const int wave = tid >> 5;
  const int m0   = blockIdx.x * 16;

  {
    const v4f* gz = (const v4f*)(zE + (size_t)m0 * CODE_DIM);
    #pragma unroll
    for (int i = 0; i < 4; ++i) {
      int q  = tid + i * 256;
      int m  = q >> 6;
      int dq = q & 63;
      *(v4f*)&sZ[m * F32_LDS_STRIDE + dq * 4] = gz[q];
    }
  }
  __syncthreads();

  const int mLane  = lane & 15;
  const int koff   = (lane >> 4) << 1;
  const int n0base = wave * 128;

  v8f acc[8];
  v8f zero = {};
  #pragma unroll
  for (int j = 0; j < 8; ++j) acc[j] = zero;

  const float* aBase = &sZ[mLane * F32_LDS_STRIDE + koff];
  for (int kk = 0; kk < 64; ++kk) {
    v2f a = *(const v2f*)(aBase + kk * 4);
    int d = kk * 4 + koff;
    #pragma unroll
    for (int j = 0; j < 8; ++j) {
      v2f b = *(const v2f*)(cb + (size_t)(n0base + j * 16 + mLane) * CODE_DIM + d);
      acc[j] = __builtin_amdgcn_wmma_f32_16x16x4_f32(
          false, a, false, b, (short)0, acc[j], false, false);
    }
  }

  float bestV[8];
  int   bestI[8];
  #pragma unroll
  for (int r = 0; r < 8; ++r) { bestV[r] = 3.4e38f; bestI[r] = 0; }

  #pragma unroll
  for (int j = 0; j < 8; ++j) {
    int code = n0base + j * 16 + mLane;
    float cs;
    if (csq) {
      cs = csq[code];
    } else {
      cs = 0.0f;
      const float* crow = cb + (size_t)code * CODE_DIM;
      for (int d = 0; d < CODE_DIM; ++d) cs += crow[d] * crow[d];
    }
    #pragma unroll
    for (int r = 0; r < 8; ++r) {
      float s = cs - 2.0f * acc[j][r];
      if (s < bestV[r]) { bestV[r] = s; bestI[r] = code; }
    }
  }

  #pragma unroll
  for (int r = 0; r < 8; ++r) {
    float v = bestV[r];
    int   ii = bestI[r];
    #pragma unroll
    for (int off = 8; off >= 1; off >>= 1) {
      float ov = __shfl_xor(v, off, 32);
      int   oi = __shfl_xor(ii, off, 32);
      if (ov < v || (ov == v && oi < ii)) { v = ov; ii = oi; }
    }
    if (mLane == 0) {
      int row = r + ((lane >> 4) << 3);
      sMinV[wave][row] = v;
      sMinI[wave][row] = ii;
    }
  }
  __syncthreads();

  if (tid < 16) {
    float v = sMinV[0][tid];
    int   ii = sMinI[0][tid];
    #pragma unroll
    for (int w = 1; w < 8; ++w) {
      float ov = sMinV[w][tid];
      int   oi = sMinI[w][tid];
      if (ov < v || (ov == v && oi < ii)) { v = ov; ii = oi; }
    }
    sFinal[tid] = ii;
    outIdx[m0 + tid] = (float)ii;
  }
  __syncthreads();

  #pragma unroll
  for (int rr = 0; rr < 2; ++rr) {
    int row  = wave * 2 + rr;
    int code = sFinal[row];
    const v4f* src = (const v4f*)(cb + (size_t)code * CODE_DIM);
    v4f* dst = (v4f*)(outZq + (size_t)(m0 + row) * CODE_DIM);
    dst[lane]      = src[lane];
    dst[lane + 32] = src[lane + 32];
  }
}

// ---------------------------------------------------------------------------
extern "C" void kernel_launch(void* const* d_in, const int* in_sizes, int n_in,
                              void* d_out, int out_size, void* d_ws, size_t ws_size,
                              hipStream_t stream) {
  const float* zE = (const float*)d_in[0];   // (8,4096,256) f32
  const float* cb = (const float*)d_in[1];   // (1024,256)  f32

  const int M = in_sizes[0] / CODE_DIM;      // 32768 query vectors
  float* outZq  = (float*)d_out;
  float* outIdx = outZq + (size_t)M * CODE_DIM;

  const size_t bshElems = (size_t)64 * BSH_TILE_ELEMS;       // 262144 bf16 each
  const size_t needBf   = bshElems * 2 * sizeof(unsigned short);  // hi + lo = 1 MB
  const size_t needCsq  = NUM_CODES * sizeof(float);

  if (ws_size >= needBf + needCsq) {
    unsigned short* bH = (unsigned short*)d_ws;
    unsigned short* bL = bH + bshElems;
    float* csqBuf = (float*)(bL + bshElems);
    vq_shufbf_kernel<<<64, 256, 0, stream>>>(cb, bH, bL);
    vq_csq_kernel<<<NUM_CODES / 8, 256, 0, stream>>>(cb, csqBuf);
    vq_main_bf16_kernel<<<M / 32, 256, 0, stream>>>(zE, cb, bH, bL, csqBuf,
                                                    outZq, outIdx);
  } else if (ws_size >= needCsq) {
    float* csqBuf = (float*)d_ws;
    vq_csq_kernel<<<NUM_CODES / 8, 256, 0, stream>>>(cb, csqBuf);
    vq_main_f32_kernel<<<M / 16, 256, 0, stream>>>(zE, cb, csqBuf, outZq, outIdx);
  } else {
    vq_main_f32_kernel<<<M / 16, 256, 0, stream>>>(zE, cb, nullptr, outZq, outIdx);
  }
}
